// SFMGTL_61967788147147
// MI455X (gfx1250) — matrix operand order repaired and checked by reading, not compiled, not run on hardware
//
#include <hip/hip_runtime.h>

// ---------------------------------------------------------------------------
// CDNA5 (gfx1250) types for WMMA bf16
// ---------------------------------------------------------------------------
typedef __attribute__((ext_vector_type(16))) __bf16 v16bf;
typedef __attribute__((ext_vector_type(8)))  __bf16 v8bf;
typedef __attribute__((ext_vector_type(8)))  float  v8f;

union BFx16 { v16bf v; v8bf h[2]; __bf16 e[16]; };

// native f32 -> bf16 (fptrunc; lowers to HW convert on gfx1250)
__device__ __forceinline__ __bf16 f2bf(float f) { return (__bf16)f; }

// ---------------------------------------------------------------------------
// Generic tiled GEMM:  C[M,N] = op(A) @ B  (f32 in/out, bf16 WMMA, f32 accum)
// TA=false: A is [M,K].  TA=true: A is [K,M] (computes A^T @ B).
// Block: 128 threads (4 waves). Tile 64x64, K-step 32.
// Path selection is wave-uniform; edge paths use clamped loads * 0/1 mask
// (loads always executed and consumed -> no EXEC predication).
// ---------------------------------------------------------------------------
template<bool TA>
__global__ __launch_bounds__(128)
void k_gemm_wmma(const float* __restrict__ A, const float* __restrict__ Bm,
                 float* __restrict__ C, int M, int N, int K,
                 long sA, long sB, long sC, const float* __restrict__ bias)
{
  __shared__ __align__(32) __bf16 As[64][32];   // [m][k]
  __shared__ __align__(32) __bf16 Bs[64][32];   // [n][k]  (B transposed)

  const int tid  = threadIdx.x;
  const int wave = tid >> 5;
  const int lane = tid & 31;
  const int mm   = lane & 15;
  const int hf   = lane >> 4;
  const int m0 = blockIdx.y * 64;
  const int n0 = blockIdx.x * 64;
  A  += (size_t)blockIdx.z * sA;
  Bm += (size_t)blockIdx.z * sB;
  C  += (size_t)blockIdx.z * sC;

  const bool fullM = (m0 + 64 <= M);   // block-uniform
  const bool fullN = (n0 + 64 <= N);   // block-uniform

  v8f acc[4];
#pragma unroll
  for (int t = 0; t < 4; ++t)
#pragma unroll
    for (int r = 0; r < 8; ++r) acc[t][r] = 0.f;

  for (int kb = 0; kb < K; kb += 32) {
    const bool fullK = (kb + 32 <= K); // block-uniform
    // ---- stage A tile into LDS as bf16 ----
    if (TA) {
      const int kr = tid >> 2, mc = (tid & 3) * 16;
      const int k = kb + kr;
      const int kcl = (k < K) ? k : (K - 1);
      const float* Ap = A + (size_t)kcl * M;
      if (fullM && fullK) {              // uniform branch
        const float* p = Ap + m0 + mc;
        float4 q0 = *(const float4*)(p);
        float4 q1 = *(const float4*)(p + 4);
        float4 q2 = *(const float4*)(p + 8);
        float4 q3 = *(const float4*)(p + 12);
        float v[16] = {q0.x,q0.y,q0.z,q0.w, q1.x,q1.y,q1.z,q1.w,
                       q2.x,q2.y,q2.z,q2.w, q3.x,q3.y,q3.z,q3.w};
#pragma unroll
        for (int i = 0; i < 16; ++i) As[mc + i][kr] = f2bf(v[i]);
      } else {
        const float km = (k < K) ? 1.f : 0.f;
#pragma unroll
        for (int i = 0; i < 16; ++i) {
          int mr = m0 + mc + i;
          int mcl = (mr < M) ? mr : (M - 1);
          float msk = (mr < M) ? km : 0.f;
          As[mc + i][kr] = f2bf(Ap[mcl] * msk);
        }
      }
    } else {
      const int r = tid >> 1, kc = (tid & 1) * 16;
      const int mr = m0 + r;
      const float* Ap = A + (size_t)((mr < M) ? mr : (M - 1)) * K;
      if (fullM && fullK) {              // uniform branch
        const float* p = Ap + kb + kc;
        float4 q0 = *(const float4*)(p);
        float4 q1 = *(const float4*)(p + 4);
        float4 q2 = *(const float4*)(p + 8);
        float4 q3 = *(const float4*)(p + 12);
        float v[16] = {q0.x,q0.y,q0.z,q0.w, q1.x,q1.y,q1.z,q1.w,
                       q2.x,q2.y,q2.z,q2.w, q3.x,q3.y,q3.z,q3.w};
#pragma unroll
        for (int i = 0; i < 16; ++i) As[r][kc + i] = f2bf(v[i]);
      } else {
        const float rmk = (mr < M) ? 1.f : 0.f;
#pragma unroll
        for (int i = 0; i < 16; ++i) {
          int k = kb + kc + i;
          int kcl = (k < K) ? k : (K - 1);
          float msk = (k < K) ? rmk : 0.f;
          As[r][kc + i] = f2bf(Ap[kcl] * msk);
        }
      }
    }
    // ---- stage B tile (transposed) into LDS as bf16 ----
    {
      const int kr = tid >> 2, nc = (tid & 3) * 16;
      const int k = kb + kr;
      const int kcl = (k < K) ? k : (K - 1);
      const float* Bp = Bm + (size_t)kcl * N;
      if (fullN && fullK) {              // uniform branch
        const float* p = Bp + n0 + nc;
        float4 q0 = *(const float4*)(p);
        float4 q1 = *(const float4*)(p + 4);
        float4 q2 = *(const float4*)(p + 8);
        float4 q3 = *(const float4*)(p + 12);
        float v[16] = {q0.x,q0.y,q0.z,q0.w, q1.x,q1.y,q1.z,q1.w,
                       q2.x,q2.y,q2.z,q2.w, q3.x,q3.y,q3.z,q3.w};
#pragma unroll
        for (int i = 0; i < 16; ++i) Bs[nc + i][kr] = f2bf(v[i]);
      } else {
        const float km = (k < K) ? 1.f : 0.f;
#pragma unroll
        for (int i = 0; i < 16; ++i) {
          int n = n0 + nc + i;
          int ncl = (n < N) ? n : (N - 1);
          float msk = (n < N) ? km : 0.f;
          Bs[nc + i][kr] = f2bf(Bp[ncl] * msk);
        }
      }
    }
    __syncthreads();

    // A fragment (16x32 bf16): lane half hf holds K in [16hf, 16hf+16)
    BFx16 a;
    a.h[0] = *reinterpret_cast<const v8bf*>(&As[16 * wave + mm][16 * hf]);
    a.h[1] = *reinterpret_cast<const v8bf*>(&As[16 * wave + mm][16 * hf + 8]);
#pragma unroll
    for (int t = 0; t < 4; ++t) {
      // B fragment (32x16): lane holds col mm, K in [16hf, 16hf+16)
      BFx16 bf;
      bf.v = *reinterpret_cast<const v16bf*>(&Bs[16 * t + mm][16 * hf]);
      acc[t] = __builtin_amdgcn_wmma_f32_16x16x32_bf16(false, a.v, false, bf.v,
                                                       (short)0, acc[t], false, false);
    }
    __syncthreads();
  }

#pragma unroll
  for (int t = 0; t < 4; ++t) {
    const int col = n0 + 16 * t + mm;
    if (col < N) {
      float bv = bias ? bias[col] : 0.f;
#pragma unroll
      for (int r = 0; r < 8; ++r) {
        const int row = m0 + 16 * wave + r + 8 * hf;
        if (row < M) C[(size_t)row * N + col] = acc[t][r] + bv;
      }
    }
  }
}

// ---------------------------------------------------------------------------
// a_src / a_dst:  out[b,n,h] = sum_c h[b,n,h,c] * att[h,c]
// ---------------------------------------------------------------------------
__global__ void k_attvec(const float* __restrict__ h, const float* __restrict__ att,
                         float* __restrict__ out, int Hh, int CH, size_t total)
{
  size_t i = (size_t)blockIdx.x * blockDim.x + threadIdx.x;
  if (i >= total) return;
  int hh = (int)(i % Hh);
  size_t bn = i / Hh;
  const float* hp = h + bn * (size_t)(Hh * CH) + (size_t)hh * CH;
  const float* ap = att + hh * CH;
  float s = 0.f;
  for (int c = 0; c < CH; ++c) s += hp[c] * ap[c];
  out[i] = s;
}

// ---------------------------------------------------------------------------
// Build hT[b][h][c][n] (bf16, zero-padded) from h [B*N, H*CH]
// ---------------------------------------------------------------------------
__global__ void k_build_hT(const float* __restrict__ h, __bf16* __restrict__ hT,
                           int Nn, int Hh, int CH, int CPAD, int NPAD, size_t total)
{
  size_t i = (size_t)blockIdx.x * blockDim.x + threadIdx.x;
  if (i >= total) return;
  int n  = (int)(i % NPAD);
  size_t r1 = i / NPAD;
  int c  = (int)(r1 % CPAD);
  int bh = (int)(r1 / CPAD);
  int b = bh / Hh, hh = bh % Hh;
  float v = (n < Nn && c < CH) ? h[((size_t)(b * Nn + n) * Hh + hh) * CH + c] : 0.f;
  hT[i] = f2bf(v);
}

// ---------------------------------------------------------------------------
// Fused GAT attention: masked leaky-relu logits -> softmax over j -> alpha@h,
// never materializing alpha. One wave per 16 output rows per (b,head).
// Probabilities are generated directly in WMMA A-fragment layout.
// TAIL=false: N is a multiple of 32 -> no tail/clamp code at all.
// Dead rows (row>=N) use a clamped row pointer and are discarded at the store.
// ---------------------------------------------------------------------------
template<int NT, bool TAIL>
__global__ __launch_bounds__(32)
void k_gat_attn(const float* __restrict__ adj, const float* __restrict__ asrc,
                const float* __restrict__ adst, const __bf16* __restrict__ hT,
                float* __restrict__ outh, int Nn, int NPAD, int Hh, int CH)
{
  __shared__ float s_src[2048];
  __shared__ float s_rs[16];

  const int lane = threadIdx.x & 31;
  const int hf   = lane >> 4;
  const int m    = lane & 15;
  const int i0   = blockIdx.x * 16;
  const int bh   = blockIdx.y;
  const int b    = bh / Hh;
  const int hh   = bh % Hh;
  const int row  = i0 + m;
  const int rowc = (row < Nn) ? row : (Nn - 1);

  for (int j = threadIdx.x; j < Nn; j += 32)
    s_src[j] = asrc[((size_t)b * Nn + j) * Hh + hh];
  __syncthreads();

  const float ad = adst[((size_t)b * Nn + rowc) * Hh + hh];
  const float* adjrow = adj + ((size_t)b * Nn + rowc) * Nn;
  const int Nfl = TAIL ? (Nn & ~31) : Nn;   // full 32-chunks
  const int Jc  = ((Nn + 31) / 32) * 32;

  // ---- pass 1: per-row max of masked leaky logits (self loops added) ----
  float rm = -3.0e38f;
  for (int j0 = 0; j0 < Nfl; j0 += 32) {
    const float* p0 = adjrow + j0 + 8 * hf;
    const float* p1 = adjrow + j0 + 16 + 8 * hf;
    float4 q0 = *(const float4*)(p0);
    float4 q1 = *(const float4*)(p0 + 4);
    float4 q2 = *(const float4*)(p1);
    float4 q3 = *(const float4*)(p1 + 4);
    float av[16] = {q0.x,q0.y,q0.z,q0.w, q1.x,q1.y,q1.z,q1.w,
                    q2.x,q2.y,q2.z,q2.w, q3.x,q3.y,q3.z,q3.w};
#pragma unroll
    for (int e = 0; e < 16; ++e) {
      int j = j0 + ((e < 8) ? (8 * hf + e) : (16 + 8 * hf + (e - 8)));
      float l = ad + s_src[j];
      l = (l > 0.f) ? l : 0.2f * l;
      bool conn = (av[e] != 0.f) | (j == rowc);
      rm = fmaxf(rm, conn ? l : -3.0e38f);
    }
  }
  if (TAIL) {
    for (int j0 = Nfl; j0 < Jc; j0 += 32) {
#pragma unroll
      for (int e = 0; e < 16; ++e) {
        int j = j0 + ((e < 8) ? (8 * hf + e) : (16 + 8 * hf + (e - 8)));
        int jc = (j < Nn) ? j : (Nn - 1);
        float inb = (j < Nn) ? 1.f : 0.f;
        float avv = adjrow[jc] * inb;            // load always consumed
        float l = ad + s_src[jc];
        l = (l > 0.f) ? l : 0.2f * l;
        bool conn = (avv != 0.f) | ((j == rowc) & (j < Nn));
        rm = fmaxf(rm, conn ? l : -3.0e38f);
      }
    }
  }
  rm = fmaxf(rm, __shfl_xor(rm, 16, 32));

  // ---- pass 2: p = exp(l-max) in A-fragment layout, WMMA against hT ----
  v8f acc[NT];
#pragma unroll
  for (int t = 0; t < NT; ++t)
#pragma unroll
    for (int r = 0; r < 8; ++r) acc[t][r] = 0.f;

  float rs = 0.f;
  const __bf16* hTb = hT + (size_t)bh * (16 * NT) * NPAD;

  for (int j0 = 0; j0 < Nfl; j0 += 32) {
    __builtin_prefetch(adjrow + j0 + 512, 0, 1);
    const float* p0 = adjrow + j0 + 8 * hf;
    const float* p1 = adjrow + j0 + 16 + 8 * hf;
    float4 q0 = *(const float4*)(p0);
    float4 q1 = *(const float4*)(p0 + 4);
    float4 q2 = *(const float4*)(p1);
    float4 q3 = *(const float4*)(p1 + 4);
    float av[16] = {q0.x,q0.y,q0.z,q0.w, q1.x,q1.y,q1.z,q1.w,
                    q2.x,q2.y,q2.z,q2.w, q3.x,q3.y,q3.z,q3.w};
    BFx16 a;
#pragma unroll
    for (int e = 0; e < 16; ++e) {
      int j = j0 + ((e < 8) ? (8 * hf + e) : (16 + 8 * hf + (e - 8)));
      float l = ad + s_src[j];
      l = (l > 0.f) ? l : 0.2f * l;
      float pv = __expf(l - rm);
      bool conn = (av[e] != 0.f) | (j == rowc);
      pv = conn ? pv : 0.f;
      rs += pv;
      a.e[e] = f2bf(pv);
    }
#pragma unroll
    for (int t = 0; t < NT; ++t) {
      BFx16 bf;
      bf.h[0] = *reinterpret_cast<const v8bf*>(hTb + (size_t)(16 * t + m) * NPAD + j0 + 8 * hf);
      bf.h[1] = *reinterpret_cast<const v8bf*>(hTb + (size_t)(16 * t + m) * NPAD + j0 + 16 + 8 * hf);
      acc[t] = __builtin_amdgcn_wmma_f32_16x16x32_bf16(false, a.v, false, bf.v,
                                                       (short)0, acc[t], false, false);
    }
  }
  if (TAIL) {
    for (int j0 = Nfl; j0 < Jc; j0 += 32) {
      BFx16 a;
#pragma unroll
      for (int e = 0; e < 16; ++e) {
        int j = j0 + ((e < 8) ? (8 * hf + e) : (16 + 8 * hf + (e - 8)));
        int jc = (j < Nn) ? j : (Nn - 1);
        float inb = (j < Nn) ? 1.f : 0.f;
        float avv = adjrow[jc] * inb;
        float l = ad + s_src[jc];
        l = (l > 0.f) ? l : 0.2f * l;
        float pv = __expf(l - rm);
        bool conn = (avv != 0.f) | ((j == rowc) & (j < Nn));
        pv = conn ? pv : 0.f;
        rs += pv;
        a.e[e] = f2bf(pv);
      }
#pragma unroll
      for (int t = 0; t < NT; ++t) {
        BFx16 bf;
        bf.h[0] = *reinterpret_cast<const v8bf*>(hTb + (size_t)(16 * t + m) * NPAD + j0 + 8 * hf);
        bf.h[1] = *reinterpret_cast<const v8bf*>(hTb + (size_t)(16 * t + m) * NPAD + j0 + 16 + 8 * hf);
        acc[t] = __builtin_amdgcn_wmma_f32_16x16x32_bf16(false, a.v, false, bf.v,
                                                         (short)0, acc[t], false, false);
      }
    }
  }
  rs += __shfl_xor(rs, 16, 32);
  if (lane < 16) s_rs[lane] = rs;
  __syncthreads();

#pragma unroll
  for (int r = 0; r < 8; ++r) {
    const int orow = i0 + r + 8 * hf;
    if (orow < Nn) {
      const float inv = 1.f / s_rs[r + 8 * hf];
#pragma unroll
      for (int t = 0; t < NT; ++t) {
        const int c = 16 * t + m;
        if (c < CH)
          outh[((size_t)bh * Nn + orow) * (16 * NT) + c] = acc[t][r] * inv;
      }
    }
  }
}

// ---------------------------------------------------------------------------
// head mean + bias:  out[b,n,c] = mean_h outh[b,h,n,c] + bias[c]
// ---------------------------------------------------------------------------
__global__ void k_headmean(const float* __restrict__ outh, const float* __restrict__ gbias,
                           float* __restrict__ out, int Nn, int Hh, int CH, int CPAD,
                           size_t total)
{
  size_t i = (size_t)blockIdx.x * blockDim.x + threadIdx.x;
  if (i >= total) return;
  int c = (int)(i % CH);
  size_t r1 = i / CH;
  int n = (int)(r1 % Nn);
  int b = (int)(r1 / Nn);
  float s = 0.f;
  for (int hh = 0; hh < Hh; ++hh)
    s += outh[((size_t)(b * Hh + hh) * Nn + n) * CPAD + c];
  out[i] = s / (float)Hh + gbias[c];
}

// ---------------------------------------------------------------------------
// GraphNorm: stats (mean, invstd of x - ms*mean over node axis) + apply
// ---------------------------------------------------------------------------
__global__ void k_gn_stats(const float* __restrict__ x, const float* __restrict__ ms,
                           float* __restrict__ aOut, float* __restrict__ iOut,
                           int Nn, int C)
{
  int b = blockIdx.y;
  int c = blockIdx.x * 64 + threadIdx.x;
  if (c >= C) return;
  const float* xp = x + (size_t)b * Nn * C + c;
  float s1 = 0.f, s2 = 0.f;
  for (int n = 0; n < Nn; ++n) {
    float v = xp[(size_t)n * C];
    s1 += v; s2 += v * v;
  }
  float mean = s1 / (float)Nn;
  float a = ms[c] * mean;
  float var = s2 / (float)Nn - 2.f * a * mean + a * a;
  aOut[b * C + c] = a;
  iOut[b * C + c] = rsqrtf(var + 1e-5f);
}

__global__ void k_gn_apply(const float* __restrict__ x, const float* __restrict__ aIn,
                           const float* __restrict__ iIn, const float* __restrict__ w,
                           const float* __restrict__ bia, float* __restrict__ y,
                           int Nn, int C, size_t total)
{
  size_t i = (size_t)blockIdx.x * blockDim.x + threadIdx.x;
  if (i >= total) return;
  int c = (int)(i % C);
  int b = (int)(i / ((size_t)Nn * C));
  y[i] = (x[i] - aIn[b * C + c]) * iIn[b * C + c] * w[c] + bia[c];
}

// ---------------------------------------------------------------------------
// Row softmax in place: rows of length K (one wave per row)
// ---------------------------------------------------------------------------
__global__ void k_rowsoftmax(float* __restrict__ s, int rows, int K)
{
  int wave = threadIdx.x >> 5, lane = threadIdx.x & 31;
  int row = blockIdx.x * (blockDim.x >> 5) + wave;
  if (row >= rows) return;
  float* p = s + (size_t)row * K;
  float m = -3.0e38f;
  for (int k = lane; k < K; k += 32) m = fmaxf(m, p[k]);
  for (int o = 16; o > 0; o >>= 1) m = fmaxf(m, __shfl_xor(m, o, 32));
  float sum = 0.f;
  for (int k = lane; k < K; k += 32) sum += __expf(p[k] - m);
  for (int o = 16; o > 0; o >>= 1) sum += __shfl_xor(sum, o, 32);
  float inv = 1.f / sum;
  for (int k = lane; k < K; k += 32) p[k] = __expf(p[k] - m) * inv;
}

// ---------------------------------------------------------------------------
// Fused link loss: per 16x16 tile compute (s s^T) via WMMA, accumulate
// sum((adj - s s^T)^2). Clamped row pointers; 0/1-mask multiplies keep all
// loads unconditional; out-of-range rows discarded in the guarded epilogue.
// ---------------------------------------------------------------------------
__global__ __launch_bounds__(32)
void k_link(const float* __restrict__ s, const float* __restrict__ adj,
            float* __restrict__ accum, int Nn, int Kd)
{
  const int lane = threadIdx.x & 31;
  const int hf = lane >> 4, m = lane & 15;
  const int i0 = blockIdx.x * 16, j0 = blockIdx.y * 16, b = blockIdx.z;
  const float* sb = s + (size_t)b * Nn * Kd;

  const int rr = i0 + m;
  const float* sa = sb + (size_t)((rr < Nn) ? rr : (Nn - 1)) * Kd;
  const int cc = j0 + m;
  const float* sc = sb + (size_t)((cc < Nn) ? cc : (Nn - 1)) * Kd;

  v8f acc;
#pragma unroll
  for (int r = 0; r < 8; ++r) acc[r] = 0.f;

  const int Kc = ((Kd + 31) / 32) * 32;
  for (int kb = 0; kb < Kc; kb += 32) {
    BFx16 a, bf;
#pragma unroll
    for (int e = 0; e < 16; ++e) {
      int k = kb + ((e < 8) ? (8 * hf + e) : (16 + 8 * hf + (e - 8)));
      int kcl = (k < Kd) ? k : (Kd - 1);
      float inb = (k < Kd) ? 1.f : 0.f;
      a.e[e] = f2bf(sa[kcl] * inb);
    }
#pragma unroll
    for (int e = 0; e < 16; ++e) {
      int k = kb + ((e < 8) ? (8 * hf + e) : (16 + 8 * hf + (e - 8)));
      int kcl = (k < Kd) ? k : (Kd - 1);
      float inb = (k < Kd) ? 1.f : 0.f;
      bf.e[e] = f2bf(sc[kcl] * inb);
    }
    acc = __builtin_amdgcn_wmma_f32_16x16x32_bf16(false, a.v, false, bf.v,
                                                  (short)0, acc, false, false);
  }

  float lsum = 0.f;
#pragma unroll
  for (int r = 0; r < 8; ++r) {
    int row = i0 + r + 8 * hf, col = j0 + m;
    if (row < Nn && col < Nn) {
      float d = adj[((size_t)b * Nn + row) * Nn + col] - acc[r];
      lsum += d * d;
    }
  }
  for (int o = 16; o > 0; o >>= 1) lsum += __shfl_xor(lsum, o, 32);
  if (lane == 0) atomicAdd(accum, lsum);
}

__global__ void k_ent(const float* __restrict__ s, float* __restrict__ accum, size_t total)
{
  size_t i = (size_t)blockIdx.x * blockDim.x + threadIdx.x;
  float v = 0.f;
  if (i < total) { float xv = s[i]; v = -xv * __logf(xv + 1e-15f); }
  for (int o = 16; o > 0; o >>= 1) v += __shfl_xor(v, o, 32);
  if ((threadIdx.x & 31) == 0) atomicAdd(accum, v);
}

// ---------------------------------------------------------------------------
// Output assembly kernels
// ---------------------------------------------------------------------------
__global__ void k_finalize(const float* __restrict__ sc, float* __restrict__ out,
                           int Bn, int N1, int K1)
{
  float l1 = sqrtf(sc[0]) / ((float)Bn * N1 * N1);
  float l2 = sqrtf(sc[2]) / ((float)Bn * K1 * K1);
  out[0] = l1 + l2;
  out[1] = sc[1] / ((float)Bn * N1) + sc[3] / ((float)Bn * K1);
}

__global__ void k_copy(const float* __restrict__ a, float* __restrict__ o, size_t n)
{
  size_t i = (size_t)blockIdx.x * blockDim.x + threadIdx.x;
  if (i < n) o[i] = a[i];
}

__global__ void k_ztswap(const float* __restrict__ zt, float* __restrict__ o,
                         int K1, int Tt, size_t total)
{
  size_t i = (size_t)blockIdx.x * blockDim.x + threadIdx.x;
  if (i >= total) return;
  int t = (int)(i % Tt);
  int k = (int)((i / Tt) % K1);
  int b = (int)(i / ((size_t)Tt * K1));
  o[i] = zt[((size_t)b * Tt + t) * K1 + k];
}

__global__ void k_semx(const float* __restrict__ sem, float* __restrict__ o,
                       int K2, int F, size_t total)
{
  size_t i = (size_t)blockIdx.x * blockDim.x + threadIdx.x;
  if (i >= total) return;
  int c = (int)(i % F);
  int b = (int)(i / F);
  float s = 0.f;
  for (int m = 0; m < K2; ++m) s += sem[((size_t)b * K2 + m) * F + c];
  o[i] = s / (float)K2;
}

__global__ void k_semt(const float* __restrict__ s2, const float* __restrict__ zt,
                       float* __restrict__ o, int K1, int K2, int Tt, size_t total)
{
  size_t i = (size_t)blockIdx.x * blockDim.x + threadIdx.x;
  if (i >= total) return;
  int mm = (int)(i % K2);
  int b = (int)(i / K2);
  float acc = 0.f;
  for (int t = 0; t < Tt; ++t)
    for (int k = 0; k < K1; ++k)
      acc += s2[((size_t)b * K1 + k) * K2 + mm] * zt[((size_t)b * Tt + t) * K1 + k];
  o[i] = acc / (float)Tt;
}

__global__ void k_qs(const float* __restrict__ zonex, const float* __restrict__ semx,
                     float* __restrict__ o, int K1, int F, size_t total)
{
  size_t i = (size_t)blockIdx.x * blockDim.x + threadIdx.x;
  if (i >= total) return;
  int c = (int)(i % F);
  int r = (int)((i / F) % (K1 + 1));
  int b = (int)(i / ((size_t)(K1 + 1) * F));
  o[i] = (r < K1) ? zonex[((size_t)b * K1 + r) * F + c] : semx[(size_t)b * F + c];
}

// ---------------------------------------------------------------------------
// Host-side orchestration helpers
// ---------------------------------------------------------------------------
static inline void gemm(hipStream_t st, bool ta, const float* A, const float* Bm, float* C,
                        int M, int N, int K, int batch, long sA, long sB, long sC,
                        const float* bias)
{
  dim3 g((unsigned)((N + 63) / 64), (unsigned)((M + 63) / 64), (unsigned)batch);
  dim3 b(128);
  if (ta) k_gemm_wmma<true ><<<g, b, 0, st>>>(A, Bm, C, M, N, K, sA, sB, sC, bias);
  else    k_gemm_wmma<false><<<g, b, 0, st>>>(A, Bm, C, M, N, K, sA, sB, sC, bias);
}

template<int NT>
static void launch_attn(hipStream_t st, dim3 g, bool tail, const float* adj,
                        const float* as, const float* ad, const __bf16* hT, float* oh,
                        int Nn, int NPAD, int Hh, int CH)
{
  if (tail) k_gat_attn<NT, true ><<<g, 32, 0, st>>>(adj, as, ad, hT, oh, Nn, NPAD, Hh, CH);
  else      k_gat_attn<NT, false><<<g, 32, 0, st>>>(adj, as, ad, hT, oh, Nn, NPAD, Hh, CH);
}

static void run_gat(hipStream_t st, const float* xin, const float* adj,
                    const float* W, const float* attS, const float* attD,
                    const float* gbias, float* out,
                    int Bn, int Nn, int cin, int CH, int NPAD,
                    float* wH, __bf16* wHT, float* wOutH, float* wSrc, float* wDst)
{
  const int Hh = 2;
  const int NT = (CH + 15) / 16;
  const int CPAD = 16 * NT;
  gemm(st, false, xin, W, wH, Bn * Nn, Hh * CH, cin, 1, 0, 0, 0, nullptr);
  {
    size_t tot = (size_t)Bn * Nn * Hh;
    int g = (int)((tot + 255) / 256);
    k_attvec<<<g, 256, 0, st>>>(wH, attS, wSrc, Hh, CH, tot);
    k_attvec<<<g, 256, 0, st>>>(wH, attD, wDst, Hh, CH, tot);
  }
  {
    size_t tot = (size_t)Bn * Hh * CPAD * NPAD;
    k_build_hT<<<(int)((tot + 255) / 256), 256, 0, st>>>(wH, wHT, Nn, Hh, CH, CPAD, NPAD, tot);
  }
  {
    dim3 g((unsigned)((Nn + 15) / 16), (unsigned)(Bn * Hh));
    const bool tail = (Nn & 31) != 0;
    switch (NT) {
      case 1:  launch_attn<1>(st, g, tail, adj, wSrc, wDst, wHT, wOutH, Nn, NPAD, Hh, CH); break;
      case 4:  launch_attn<4>(st, g, tail, adj, wSrc, wDst, wHT, wOutH, Nn, NPAD, Hh, CH); break;
      case 7:  launch_attn<7>(st, g, tail, adj, wSrc, wDst, wHT, wOutH, Nn, NPAD, Hh, CH); break;
      default: launch_attn<8>(st, g, tail, adj, wSrc, wDst, wHT, wOutH, Nn, NPAD, Hh, CH); break;
    }
  }
  {
    size_t tot = (size_t)Bn * Nn * CH;
    k_headmean<<<(int)((tot + 255) / 256), 256, 0, st>>>(wOutH, gbias, out, Nn, Hh, CH, CPAD, tot);
  }
}

static void run_gn(hipStream_t st, float* xio, const float* wgt, const float* bia,
                   const float* ms, int Bn, int Nn, int C, float* wGA, float* wGI)
{
  dim3 g((unsigned)((C + 63) / 64), (unsigned)Bn);
  k_gn_stats<<<g, 64, 0, st>>>(xio, ms, wGA, wGI, Nn, C);
  size_t tot = (size_t)Bn * Nn * C;
  k_gn_apply<<<(int)((tot + 255) / 256), 256, 0, st>>>(xio, wGA, wGI, wgt, bia, xio, Nn, C, tot);
}

static void run_gnn(hipStream_t st, const float* const* P, const float* xin,
                    const float* adj, float* out, int Bn, int Nn, int cin, int chid,
                    int cout, int NPAD, float* wH, __bf16* wHT, float* wOutH,
                    float* wSrc, float* wDst, float* wX1, float* wGA, float* wGI)
{
  run_gat(st, xin, adj, P[0], P[1], P[2], P[3], wX1, Bn, Nn, cin, chid, NPAD,
          wH, wHT, wOutH, wSrc, wDst);
  run_gn(st, wX1, P[4], P[5], P[6], Bn, Nn, chid, wGA, wGI);
  run_gat(st, wX1, adj, P[7], P[8], P[9], P[10], out, Bn, Nn, chid, cout, NPAD,
          wH, wHT, wOutH, wSrc, wDst);
  run_gn(st, out, P[11], P[12], P[13], Bn, Nn, cout, wGA, wGI);
}

static inline char* wsal(char*& p, size_t bytes) {
  char* r = p;
  p += (bytes + 255) & ~(size_t)255;
  return r;
}

// ---------------------------------------------------------------------------
extern "C" void kernel_launch(void* const* d_in, const int* in_sizes, int n_in,
                              void* d_out, int out_size, void* d_ws, size_t ws_size,
                              hipStream_t stream)
{
  (void)in_sizes; (void)n_in; (void)out_size; (void)ws_size;
  constexpr int Bn = 4, N1 = 2048, F = 64, HID = 128, K1 = 100, K2 = 10, Tt = 12;

  const float* x    = (const float*)d_in[0];
  const float* adj  = (const float*)d_in[1];
  const float* oseq = (const float*)d_in[2];
  const float* P[4][14];
  for (int g = 0; g < 4; ++g)
    for (int k = 0; k < 14; ++k)
      P[g][k] = (const float*)d_in[3 + g * 14 + k];

  char* wp = (char*)d_ws;
  float*  wH    = (float*) wsal(wp, (size_t)Bn * N1 * 256 * 4);        // h (max)
  __bf16* wHT   = (__bf16*)wsal(wp, (size_t)Bn * 2 * 128 * N1 * 2);    // hT bf16
  float*  wOutH = (float*) wsal(wp, (size_t)Bn * 2 * N1 * 128 * 4);    // per-head out
  float*  wSrc  = (float*) wsal(wp, (size_t)Bn * N1 * 2 * 4);
  float*  wDst  = (float*) wsal(wp, (size_t)Bn * N1 * 2 * 4);
  float*  wX1   = (float*) wsal(wp, (size_t)Bn * N1 * HID * 4);
  float*  wGA   = (float*) wsal(wp, (size_t)Bn * 128 * 4);
  float*  wGI   = (float*) wsal(wp, (size_t)Bn * 128 * 4);
  float*  wS1   = (float*) wsal(wp, (size_t)Bn * N1 * K1 * 4);
  float*  wNX   = (float*) wsal(wp, (size_t)Bn * N1 * F * 4);
  float*  wTmp  = (float*) wsal(wp, (size_t)Bn * N1 * K1 * 4);
  float*  wAdjZ = (float*) wsal(wp, (size_t)Bn * K1 * K1 * 4);
  float*  wZX   = (float*) wsal(wp, (size_t)Bn * K1 * F * 4);
  float*  wS2   = (float*) wsal(wp, (size_t)Bn * K1 * K2 * 4);
  float*  wZ2   = (float*) wsal(wp, (size_t)Bn * K1 * F * 4);
  float*  wSem  = (float*) wsal(wp, (size_t)Bn * K2 * F * 4);
  float*  wZT   = (float*) wsal(wp, (size_t)Bn * Tt * K1 * 4);
  float*  wScal = (float*) wsal(wp, 4 * 4);
  float*  out   = (float*)d_out;

  // ---- level 1 GNNs on N=2048 ----
  run_gnn(stream, P[0], x, adj, wS1, Bn, N1, F, HID, K1, 2048,
          wH, wHT, wOutH, wSrc, wDst, wX1, wGA, wGI);
  run_gnn(stream, P[1], x, adj, wNX, Bn, N1, F, HID, F, 2048,
          wH, wHT, wOutH, wSrc, wDst, wX1, wGA, wGI);

  // ---- diff-pool 1 ----
  k_rowsoftmax<<<(Bn * N1 + 7) / 8, 256, 0, stream>>>(wS1, Bn * N1, K1);
  gemm(stream, true,  wS1, wNX,  wZX,   K1, F,  N1, Bn, (long)N1 * K1, (long)N1 * F,  (long)K1 * F,  nullptr);
  gemm(stream, false, adj, wS1,  wTmp,  N1, K1, N1, Bn, (long)N1 * N1, (long)N1 * K1, (long)N1 * K1, nullptr);
  gemm(stream, true,  wS1, wTmp, wAdjZ, K1, K1, N1, Bn, (long)N1 * K1, (long)N1 * K1, (long)K1 * K1, nullptr);

  hipMemsetAsync(wScal, 0, 4 * sizeof(float), stream);
  {
    dim3 g((unsigned)((N1 + 15) / 16), (unsigned)((N1 + 15) / 16), (unsigned)Bn);
    k_link<<<g, 32, 0, stream>>>(wS1, adj, &wScal[0], N1, K1);
    size_t tot = (size_t)Bn * N1 * K1;
    k_ent<<<(int)((tot + 255) / 256), 256, 0, stream>>>(wS1, &wScal[1], tot);
  }
  // zone_temp = origin_seq @ softmax(s1)  -> [B,T,K1]
  gemm(stream, false, oseq, wS1, wZT, Tt, K1, N1, Bn, (long)Tt * N1, (long)N1 * K1, (long)Tt * K1, nullptr);

  // ---- level 2 GNNs on K1=100 nodes ----
  run_gnn(stream, P[2], wZX, wAdjZ, wS2, Bn, K1, F, HID, K2, 128,
          wH, wHT, wOutH, wSrc, wDst, wX1, wGA, wGI);
  run_gnn(stream, P[3], wZX, wAdjZ, wZ2, Bn, K1, F, HID, F, 128,
          wH, wHT, wOutH, wSrc, wDst, wX1, wGA, wGI);

  // ---- diff-pool 2 ----
  k_rowsoftmax<<<(Bn * K1 + 7) / 8, 256, 0, stream>>>(wS2, Bn * K1, K2);
  gemm(stream, true, wS2, wZ2, wSem, K2, F, K1, Bn, (long)K1 * K2, (long)K1 * F, (long)K2 * F, nullptr);
  {
    dim3 g((unsigned)((K1 + 15) / 16), (unsigned)((K1 + 15) / 16), (unsigned)Bn);
    k_link<<<g, 32, 0, stream>>>(wS2, wAdjZ, &wScal[2], K1, K2);
    size_t tot = (size_t)Bn * K1 * K2;
    k_ent<<<(int)((tot + 255) / 256), 256, 0, stream>>>(wS2, &wScal[3], tot);
  }

  // ---- assemble outputs ----
  const int offZX = 2;
  const int offZT = offZX + Bn * K1 * F;
  const int offSX = offZT + Bn * K1 * Tt;
  const int offST = offSX + Bn * F;
  const int offQS = offST + Bn * K2;

  k_finalize<<<1, 1, 0, stream>>>(wScal, out, Bn, N1, K1);
  {
    size_t n = (size_t)Bn * K1 * F;
    k_copy<<<(int)((n + 255) / 256), 256, 0, stream>>>(wZX, out + offZX, n);
  }
  {
    size_t n = (size_t)Bn * K1 * Tt;
    k_ztswap<<<(int)((n + 255) / 256), 256, 0, stream>>>(wZT, out + offZT, K1, Tt, n);
  }
  {
    size_t n = (size_t)Bn * F;
    k_semx<<<(int)((n + 255) / 256), 256, 0, stream>>>(wSem, out + offSX, K2, F, n);
  }
  {
    size_t n = (size_t)Bn * K2;
    k_semt<<<(int)((n + 255) / 256), 256, 0, stream>>>(wS2, wZT, out + offST, K1, K2, Tt, n);
  }
  {
    size_t n = (size_t)Bn * (K1 + 1) * F;
    k_qs<<<(int)((n + 255) / 256), 256, 0, stream>>>(wZX, out + offSX, out + offQS, K1, F, n);
  }
}
